// LatentTemplateCRFAR_36859409334320
// MI455X (gfx1250) — matrix operand order repaired and checked by reading, not compiled, not run on hardware
//
#include <hip/hip_runtime.h>

#define V_  32000
#define K_  32
#define D_  256
#define T_  128
#define B_  128
#define M_  16
#define D4_ 1024

typedef __attribute__((ext_vector_type(16))) __bf16 bf16x16;
typedef __attribute__((ext_vector_type(8)))  float  f32x8;

union FragU { bf16x16 v; unsigned u[8]; unsigned short h[16]; };

__device__ __forceinline__ unsigned short f2bf(float x) {
  unsigned u = __float_as_uint(x);
  unsigned r = (u + 0x7FFFu + ((u >> 16) & 1u)) >> 16;
  return (unsigned short)r;
}
__device__ __forceinline__ float sigf(float x) { return 1.f / (1.f + __expf(-x)); }

// ---- WMMA fragment loaders (16x16x32 bf16, contiguous k = 16*half + s in both A and B) ----
__device__ __forceinline__ bf16x16 ldfragA(const unsigned short* A, int lda, int m0, int k0, int lane) {
  FragU f;
  int m = m0 + (lane & 15);
  int half = lane >> 4;
  const unsigned short* p = A + (size_t)m * lda + k0;
#pragma unroll
  for (int v = 0; v < 8; ++v) f.u[v] = *(const unsigned*)(p + half * 16 + 2 * v);
  return f.v;
}
__device__ __forceinline__ bf16x16 ldfragB(const unsigned short* Bt, int ldb, int n0, int k0, int lane) {
  FragU f;
  int n = n0 + (lane & 15);
  int half = lane >> 4;
  const unsigned short* p = Bt + (size_t)n * ldb + k0 + half * 16;
#pragma unroll
  for (int v = 0; v < 8; ++v) f.u[v] = *(const unsigned*)(p + 2 * v);
  return f.v;
}
__device__ __forceinline__ f32x8 wmma_bf16(bf16x16 a, bf16x16 b, f32x8 c) {
  return __builtin_amdgcn_wmma_f32_16x16x32_bf16(false, a, false, b, (short)0, c, false, false);
}

// ================= weight prep =================
__global__ void prep_comb_w(const float* __restrict__ Wih, const float* __restrict__ Whh,
                            unsigned short* __restrict__ Wt) {
  int idx = blockIdx.x * blockDim.x + threadIdx.x; // 1024*512
  int k = idx & 511, n = idx >> 9;
  float v = (k < 256) ? Wih[k * D4_ + n] : Whh[(k - 256) * D4_ + n];
  Wt[n * 512 + k] = f2bf(v);
}
__global__ void prep_outw(const float* __restrict__ W, unsigned short* __restrict__ Wt) {
  int idx = blockIdx.x * blockDim.x + threadIdx.x; // 32000*256
  int k = idx & 255, n = idx >> 8;
  Wt[(size_t)n * 256 + k] = f2bf(W[(size_t)k * V_ + n]);
}

// ================= embedding gather =================
__global__ void gather_sent(const int* __restrict__ sentences, const float* __restrict__ emb,
                            float* __restrict__ sent_emb) {
  int idx = blockIdx.x, d = threadIdx.x; // idx = b*T + t
  int w = sentences[idx];
  sent_emb[(size_t)idx * D_ + d] = emb[(size_t)w * D_ + d];
}

__global__ void enc_init(const float* __restrict__ sent_emb, unsigned short* __restrict__ A_enc,
                         float* __restrict__ c_enc, float* __restrict__ rowsum, float* __restrict__ accs) {
  int b = blockIdx.x, d = threadIdx.x;
  A_enc[b * 512 + d] = f2bf(sent_emb[((size_t)b * T_) * D_ + d]);
  A_enc[b * 512 + 256 + d] = 0;
  c_enc[b * D_ + d] = 0.f;
  if (d == 0) rowsum[b] = 0.f;
  if (b == 0 && d < 2) accs[d] = 0.f;
}

// ===== generic WMMA GEMM (small recurrent GEMMs): C[m][n] = A[m][:K]*Bt[n][:K] (+bias) =====
__global__ void gemm_bf16(const unsigned short* __restrict__ A, int lda,
                          const unsigned short* __restrict__ Bt, int ldb,
                          float* __restrict__ C, int ldc,
                          const float* __restrict__ bias, int K) {
  int lane = threadIdx.x & 31, w = threadIdx.x >> 5;
  int m0 = blockIdx.y * 128 + w * 16;
  int n0 = blockIdx.x * 16;
  f32x8 c = {};
  for (int k0 = 0; k0 < K; k0 += 32) {
    bf16x16 a = ldfragA(A, lda, m0, k0, lane);
    bf16x16 b = ldfragB(Bt, ldb, n0, k0, lane);
    c = wmma_bf16(a, b, c);
  }
  int n = n0 + (lane & 15), half = lane >> 4;
#pragma unroll
  for (int v = 0; v < 8; ++v) {
    int m = m0 + v + 8 * half;
    float val = c[v];
    if (bias) val += bias[n];
    C[(size_t)m * ldc + n] = val;
  }
}

// ================= LSTM pointwise =================
__global__ void lstm_pw_enc(int t, const float* __restrict__ gates, float* __restrict__ c_enc,
                            float* __restrict__ enc_out, const float* __restrict__ sent_emb,
                            unsigned short* __restrict__ A_enc) {
  int b = blockIdx.x, d = threadIdx.x;
  const float* g = gates + (size_t)b * D4_;
  float gi = g[d], gf = g[256 + d], gg = g[512 + d], go = g[768 + d];
  float c = c_enc[b * D_ + d];
  c = sigf(gf) * c + sigf(gi) * tanhf(gg);
  float h = sigf(go) * tanhf(c);
  c_enc[b * D_ + d] = c;
  enc_out[((size_t)t * B_ + b) * D_ + d] = h;
  A_enc[b * 512 + 256 + d] = f2bf(h);
  if (t + 1 < T_) A_enc[b * 512 + d] = f2bf(sent_emb[((size_t)b * T_ + t + 1) * D_ + d]);
}

__global__ void lstm_pw_dec(int t, const float* __restrict__ gates, float* __restrict__ c_dec,
                            float* __restrict__ h_dec, const unsigned short* __restrict__ dec_in,
                            unsigned short* __restrict__ A_dec) {
  int b = blockIdx.x, d = threadIdx.x;
  const float* g = gates + (size_t)b * D4_;
  float gi = g[d], gf = g[256 + d], gg = g[512 + d], go = g[768 + d];
  float c = c_dec[b * D_ + d];
  c = sigf(gf) * c + sigf(gi) * tanhf(gg);
  float h = sigf(go) * tanhf(c);
  c_dec[b * D_ + d] = c;
  h_dec[b * D_ + d] = h;
  A_dec[b * 512 + 256 + d] = f2bf(h);
  if (t + 1 <= T_ - 2) A_dec[b * 512 + d] = dec_in[((size_t)b * (T_ - 1) + t + 1) * D_ + d];
}

// ================= phi = enc_out @ crf_W + b, then temporal smoothing ==========
__global__ void phi_raw_k(const float* __restrict__ enc_out, const float* __restrict__ crf_W,
                          const float* __restrict__ crf_b, float* __restrict__ praw) {
  int idx = blockIdx.x * blockDim.x + threadIdx.x; // T*B*K
  int k = idx & (K_ - 1);
  int tb = idx >> 5;
  const float* e = enc_out + (size_t)tb * D_;
  float s = crf_b[k];
  for (int d = 0; d < D_; ++d) s += e[d] * crf_W[d * K_ + k];
  praw[idx] = s;
}
__global__ void phi_smooth_k(const float* __restrict__ praw, float* __restrict__ phiT) {
  int idx = blockIdx.x * blockDim.x + threadIdx.x;
  int k = idx & 31;
  int tb = idx >> 5;
  int b = tb % B_, t = tb / B_;
  float v = praw[idx];
  if (t < T_ - 1) v += praw[(((t + 1) * B_) + b) * K_ + k];
  if (t > 0) v += praw[(((t - 1) * B_) + b) * K_ + k] + praw[((t * B_) + b) * K_ + k];
  phiT[idx] = v;
}

// ================= CRF forward (one wave32 per batch row) =================
__global__ void crf_fwd(const float* __restrict__ phiT, const float* __restrict__ trans,
                        const int* __restrict__ lens, float* __restrict__ alphas,
                        float* __restrict__ ent_acc) {
  int b = blockIdx.x, j = threadIdx.x;
  __shared__ float sT[K_ * K_], sa[K_], sH[K_], tmp[K_];
  for (int i = j; i < K_ * K_; i += K_) sT[i] = trans[i];
  float a0 = phiT[(0 * B_ + b) * K_ + j];
  sa[j] = a0; sH[j] = 0.f;
  alphas[(0 * B_ + b) * K_ + j] = a0;
  int L = lens[b];
  __syncthreads();
  for (int t = 1; t < T_; ++t) {
    float mx = -1e30f;
    for (int i = 0; i < K_; ++i) mx = fmaxf(mx, sa[i] + sT[i * K_ + j]);
    float s = 0.f;
    for (int i = 0; i < K_; ++i) s += __expf(sa[i] + sT[i * K_ + j] - mx);
    float lZ = mx + __logf(s);
    float nH = 0.f;
    for (int i = 0; i < K_; ++i) {
      float lw = sa[i] + sT[i * K_ + j] - lZ;
      nH += __expf(lw) * (sH[i] - lw);
    }
    float na = lZ + phiT[((size_t)t * B_ + b) * K_ + j];
    bool act = t < L;
    float aj = act ? na : sa[j];
    float Hj = act ? nH : sH[j];
    __syncthreads();
    sa[j] = aj; sH[j] = Hj;
    alphas[((size_t)t * B_ + b) * K_ + j] = aj;
    __syncthreads();
  }
  float mx = -1e30f;
  for (int i = 0; i < K_; ++i) mx = fmaxf(mx, sa[i]);
  float s = 0.f;
  for (int i = 0; i < K_; ++i) s += __expf(sa[i] - mx);
  float lZ = mx + __logf(s);
  float lp = sa[j] - lZ;
  tmp[j] = __expf(lp) * (sH[j] - lp);
  __syncthreads();
  if (j == 0) {
    float e = 0.f;
    for (int i = 0; i < K_; ++i) e += tmp[i];
    atomicAdd(ent_acc, e);
  }
}

// ================= Gumbel backward sampling (one wave32 per batch row) ==========
__global__ void crf_bwd(const float* __restrict__ alphas, const float* __restrict__ trans,
                        const float* __restrict__ gumbel, const int* __restrict__ lens,
                        const float* __restrict__ tau_p, float* __restrict__ zsample,
                        int* __restrict__ z_ids) {
  int b = blockIdx.x, j = threadIdx.x;
  __shared__ float sT[K_ * K_], sl[K_], tmp[K_];
  __shared__ int snid;
  for (int i = j; i < K_ * K_; i += K_) sT[i] = trans[i];
  if (j == 0) snid = 0;
  float tau = tau_p[0];
  int L = lens[b];
  __syncthreads();
  for (int t = T_ - 1; t >= 0; --t) {
    int nid = snid;
    float a = alphas[((size_t)t * B_ + b) * K_ + j];
    bool is_last = (t == L - 1);
    bool active = (t < L);
    float logit = is_last ? a : a + sT[j * K_ + nid];
    float g = gumbel[((size_t)t * B_ + b) * K_ + j];
    sl[j] = (logit + g) / tau;
    __syncthreads();
    float mx = -1e30f;
    for (int i = 0; i < K_; ++i) mx = fmaxf(mx, sl[i]);
    float s = 0.f;
    for (int i = 0; i < K_; ++i) s += __expf(sl[i] - mx);
    float y = active ? __expf(sl[j] - mx) / s : 0.f;
    zsample[((size_t)b * T_ + t) * K_ + j] = y;
    tmp[j] = y;
    __syncthreads();
    if (j == 0) {
      int best = 0; float bv = tmp[0];
      for (int i = 1; i < K_; ++i) if (tmp[i] > bv) { bv = tmp[i]; best = i; }
      int ids = active ? best : 0;
      z_ids[b * T_ + t] = ids;
      snid = ids;
    }
    __syncthreads();
  }
}

// ========== zse = zsample @ z_emb; build dec inputs + z_next ==========
__global__ void build_dec_inputs(const float* __restrict__ zsample, const float* __restrict__ z_emb,
                                 const float* __restrict__ sent_emb, const float* __restrict__ drop_unif,
                                 const float* __restrict__ xl_p, unsigned short* __restrict__ dec_in,
                                 float* __restrict__ z_next, unsigned short* __restrict__ A_dec) {
  int idx = blockIdx.x; // b*T + t
  int b = idx / T_, t = idx % T_, d = threadIdx.x;
  float zr = 0.f;
  const float* zs = zsample + (size_t)idx * K_;
  for (int k = 0; k < K_; ++k) zr += zs[k] * z_emb[k * D_ + d];
  if (t >= 1) z_next[((size_t)b * (T_ - 1) + (t - 1)) * D_ + d] = zr;
  if (t <= T_ - 2) {
    float wm = (drop_unif[b * T_ + t] > xl_p[0]) ? 1.f : 0.f;
    float di = ((t == 0) ? 0.f : zr) + sent_emb[(size_t)idx * D_ + d] * wm;
    unsigned short h = f2bf(di);
    dec_in[((size_t)b * (T_ - 1) + t) * D_ + d] = h;
    if (t == 0) A_dec[b * 512 + d] = h;
  }
}

// ================= KV memory + decoder init state =================
__global__ void kv_setup(const int* __restrict__ keys, const int* __restrict__ vals,
                         const float* __restrict__ emb, const float* __restrict__ init_Wh,
                         const float* __restrict__ init_bh, const float* __restrict__ init_Wc,
                         const float* __restrict__ init_bc, const float* __restrict__ attn_Wk,
                         const float* __restrict__ cattn_Wk, float* __restrict__ kv_emb,
                         float* __restrict__ memk, float* __restrict__ memk2,
                         float* __restrict__ h_dec, float* __restrict__ c_dec,
                         unsigned short* __restrict__ A_dec) {
  int b = blockIdx.x, d = threadIdx.x;
  __shared__ float ske[M_ * D_];
  __shared__ float skenc[D_];
  float cnt = 0.f, acc = 0.f;
  for (int m = 0; m < M_; ++m) {
    int kk = keys[b * M_ + m], vv = vals[b * M_ + m];
    float e = emb[(size_t)kk * D_ + d] + emb[(size_t)vv * D_ + d];
    ske[m * D_ + d] = e;
    kv_emb[((size_t)b * M_ + m) * D_ + d] = e;
    if (kk != 0) { cnt += 1.f; acc += e; }
  }
  skenc[d] = acc / cnt;
  __syncthreads();
  float h0 = init_bh[d], c0 = init_bc[d];
  for (int k = 0; k < D_; ++k) {
    float kv = skenc[k];
    h0 += kv * init_Wh[k * D_ + d];
    c0 += kv * init_Wc[k * D_ + d];
  }
  h_dec[b * D_ + d] = h0;
  c_dec[b * D_ + d] = c0;
  A_dec[b * 512 + 256 + d] = f2bf(h0);
  for (int m = 0; m < M_; ++m) {
    float s1 = 0.f, s2 = 0.f;
    for (int k = 0; k < D_; ++k) {
      float e = ske[m * D_ + k];
      s1 += e * attn_Wk[k * D_ + d];
      s2 += e * cattn_Wk[k * D_ + d];
    }
    memk[((size_t)b * M_ + m) * D_ + d] = s1;
    memk2[((size_t)b * M_ + m) * D_ + d] = s2;
  }
}

// ===== fused per-row decoder small ops: attention, dec_out, lpz, dec_int, g, copy, logit_tx =====
__global__ void dec_fused_small(int t, const float* __restrict__ h_dec, const float* __restrict__ kv_emb,
                                const float* __restrict__ memk, const float* __restrict__ memk2,
                                const int* __restrict__ keys, const float* __restrict__ attn_Wq,
                                const float* __restrict__ attn_v, const float* __restrict__ attn_out_W,
                                const float* __restrict__ attn_out_b, const float* __restrict__ pz_W,
                                const float* __restrict__ pz_b, const float* __restrict__ int_W,
                                const float* __restrict__ int_b, const float* __restrict__ cattn_Wq,
                                const float* __restrict__ cattn_v, const float* __restrict__ g_W,
                                const float* __restrict__ g_b, const float* __restrict__ out_W,
                                const float* __restrict__ out_b, const float* __restrict__ z_next,
                                const int* __restrict__ sentences, const int* __restrict__ z_ids,
                                unsigned short* __restrict__ decint_bf16, float* __restrict__ scal) {
  int b = blockIdx.x, tid = threadIdx.x;
  __shared__ float sh[D_], sq[D_], sctx[D_], sdo[D_], szn[D_], sdi[D_], sq2[D_], spart[D_];
  __shared__ float se[M_], sa[M_], szl[K_];
  __shared__ float sg, scp, slpz;
  sh[tid] = h_dec[b * D_ + tid];
  szn[tid] = z_next[((size_t)b * (T_ - 1) + t) * D_ + tid];
  __syncthreads();
  float q = 0.f;
  for (int k = 0; k < D_; ++k) q += sh[k] * attn_Wq[k * D_ + tid];
  sq[tid] = q;
  __syncthreads();
  {
    int m = tid >> 4, j = tid & 15;
    float p = 0.f;
    const float* mk = memk + ((size_t)b * M_ + m) * D_;
    for (int k = j * 16; k < j * 16 + 16; ++k) p += attn_v[k] * tanhf(sq[k] + mk[k]);
    spart[tid] = p;
  }
  __syncthreads();
  if (tid < M_) {
    float e = 0.f;
    for (int j = 0; j < 16; ++j) e += spart[tid * 16 + j];
    se[tid] = (keys[b * M_ + tid] != 0) ? e : -1e9f;
  }
  __syncthreads();
  if (tid < M_) {
    float mx = -1e30f;
    for (int i = 0; i < M_; ++i) mx = fmaxf(mx, se[i]);
    float s = 0.f;
    for (int i = 0; i < M_; ++i) s += __expf(se[i] - mx);
    sa[tid] = __expf(se[tid] - mx) / s;
  }
  __syncthreads();
  float cx = 0.f;
  for (int m = 0; m < M_; ++m) cx += sa[m] * kv_emb[((size_t)b * M_ + m) * D_ + tid];
  sctx[tid] = cx;
  __syncthreads();
  float dv = attn_out_b[tid];
  for (int k = 0; k < D_; ++k) dv += sh[k] * attn_out_W[k * D_ + tid];
  for (int k = 0; k < D_; ++k) dv += sctx[k] * attn_out_W[(D_ + k) * D_ + tid];
  sdo[tid] = dv;
  __syncthreads();
  if (tid < K_) {
    float z = pz_b[tid];
    for (int k = 0; k < D_; ++k) z += sdo[k] * pz_W[k * K_ + tid];
    szl[tid] = z;
  }
  __syncthreads();
  if (tid == 0) {
    float mx = -1e30f;
    for (int i = 0; i < K_; ++i) mx = fmaxf(mx, szl[i]);
    float s = 0.f;
    for (int i = 0; i < K_; ++i) s += __expf(szl[i] - mx);
    int tz = z_ids[b * T_ + t + 1];
    slpz = szl[tz] - mx - __logf(s);
  }
  float di = int_b[tid];
  for (int k = 0; k < D_; ++k) di += sdo[k] * int_W[k * D_ + tid];
  for (int k = 0; k < D_; ++k) di += szn[k] * int_W[(D_ + k) * D_ + tid];
  sdi[tid] = di;
  decint_bf16[b * D_ + tid] = f2bf(di);
  __syncthreads();
  float q2 = 0.f;
  for (int k = 0; k < D_; ++k) q2 += sdi[k] * cattn_Wq[k * D_ + tid];
  sq2[tid] = q2;
  __syncthreads();
  {
    int m = tid >> 4, j = tid & 15;
    float p = 0.f;
    const float* mk = memk2 + ((size_t)b * M_ + m) * D_;
    for (int k = j * 16; k < j * 16 + 16; ++k) p += cattn_v[k] * tanhf(sq2[k] + mk[k]);
    spart[tid] = p;
  }
  __syncthreads();
  if (tid < M_) {
    float e = 0.f;
    for (int j = 0; j < 16; ++j) e += spart[tid * 16 + j];
    se[tid] = (keys[b * M_ + tid] != 0) ? e : -1e9f;
  }
  __syncthreads();
  int tx = sentences[b * T_ + t + 1];
  if (tid == 0) {
    float mx = -1e30f;
    for (int i = 0; i < M_; ++i) mx = fmaxf(mx, se[i]);
    float s = 0.f;
    for (int i = 0; i < M_; ++i) s += __expf(se[i] - mx);
    float cp = 0.f;
    for (int i = 0; i < M_; ++i) {
      float a = __expf(se[i] - mx) / s;
      if (keys[b * M_ + i] == tx) cp += a;
    }
    scp = cp;
  }
  spart[tid] = sdi[tid] * g_W[tid];
  __syncthreads();
  for (int s = 128; s > 0; s >>= 1) {
    if (tid < s) spart[tid] += spart[tid + s];
    __syncthreads();
  }
  if (tid == 0) sg = spart[0] + g_b[0];
  __syncthreads();
  spart[tid] = sdi[tid] * out_W[(size_t)tid * V_ + tx];
  __syncthreads();
  for (int s = 128; s > 0; s >>= 1) {
    if (tid < s) spart[tid] += spart[tid + s];
    __syncthreads();
  }
  if (tid == 0) {
    scal[b * 4 + 0] = sigf(sg);
    scal[b * 4 + 1] = scp;
    scal[b * 4 + 2] = spart[0] + out_b[tx];
    scal[b * 4 + 3] = slpz;
  }
}

// ===== dominant kernel: WMMA output projection fused with sum(exp(logits)) per row =====
// Block tile: M=128 (8 waves x 16 rows), N=64 (4 WMMA tiles per wave, A reused).
// B stripe (64x256 bf16 = 32KB, contiguous) staged in LDS once; K=256 fully resident.
__global__ void out_proj_sumexp(const unsigned short* __restrict__ Aint,
                                const unsigned short* __restrict__ Wt,
                                const float* __restrict__ out_b,
                                float* __restrict__ rowsum) {
  __shared__ unsigned short Bs[64 * 256]; // 32 KB
  int tid = threadIdx.x;
  int lane = tid & 31, w = tid >> 5;
  int n0 = blockIdx.x * 64;
  int m0 = w * 16;
  // L2->L0 prefetch of the A panel this wave will read (global_prefetch_b8 path)
  __builtin_prefetch(Aint + (size_t)m0 * 256, 0, 3);
  // cooperative stage of contiguous B stripe into LDS (b128 loads)
  {
    const uint4* src = (const uint4*)(Wt + (size_t)n0 * 256);
    uint4* dst = (uint4*)Bs;
#pragma unroll
    for (int i = 0; i < 8; ++i) dst[tid + i * 256] = src[tid + i * 256];
  }
  __syncthreads();
  int nl = lane & 15, half = lane >> 4;
  f32x8 cc[4] = {{}, {}, {}, {}};
#pragma unroll
  for (int k0 = 0; k0 < 256; k0 += 32) {
    bf16x16 a = ldfragA(Aint, 256, m0, k0, lane);
#pragma unroll
    for (int tt = 0; tt < 4; ++tt) {
      FragU fb;
#pragma unroll
      for (int v = 0; v < 8; ++v)
        fb.u[v] = *(const unsigned*)&Bs[(nl + tt * 16) * 256 + k0 + half * 16 + 2 * v];
      cc[tt] = wmma_bf16(a, fb.v, cc[tt]);
    }
  }
  // epilogue: per-row sum of exp over the 4 tiles, then lane-shuffle reduce + one atomic per row
  float acc[8] = {0.f, 0.f, 0.f, 0.f, 0.f, 0.f, 0.f, 0.f};
#pragma unroll
  for (int tt = 0; tt < 4; ++tt) {
    float bb = out_b[n0 + tt * 16 + nl];
#pragma unroll
    for (int v = 0; v < 8; ++v) acc[v] += __expf(cc[tt][v] + bb);
  }
#pragma unroll
  for (int v = 0; v < 8; ++v) {
    float e = acc[v];
    for (int off = 1; off < 16; off <<= 1) e += __shfl_xor(e, off, 32);
    if (nl == 0) atomicAdd(&rowsum[m0 + v + 8 * half], e);
  }
}

__global__ void dec_finalize(int t, const float* __restrict__ scal, float* __restrict__ rowsum,
                             const int* __restrict__ lens, float* __restrict__ lp_acc) {
  int b = threadIdx.x; // 128 threads
  __shared__ float sp[B_];
  float rs = rowsum[b];
  rowsum[b] = 0.f;
  float g = scal[b * 4 + 0], cp = scal[b * 4 + 1], lt = scal[b * 4 + 2], lpz = scal[b * 4 + 3];
  float p = (1.f - g) * __expf(lt) / rs + g * cp + 1e-10f;
  float v = (t < lens[b]) ? (logf(p) + lpz) : 0.f;
  sp[b] = v;
  __syncthreads();
  if (b == 0) {
    float s = 0.f;
    for (int i = 0; i < B_; ++i) s += sp[i];
    lp_acc[0] += s;
  }
}

__global__ void final_loss(const float* __restrict__ accs, const int* __restrict__ lens,
                           float* __restrict__ out) {
  if (threadIdx.x == 0) {
    float denom = 0.f;
    for (int b = 0; b < B_; ++b) denom += (float)lens[b];
    out[0] = -(accs[0] / (float)B_ + accs[1] / denom);
  }
}

// ================================================================
extern "C" void kernel_launch(void* const* d_in, const int* in_sizes, int n_in,
                              void* d_out, int out_size, void* d_ws, size_t ws_size,
                              hipStream_t stream) {
  (void)in_sizes; (void)n_in; (void)out_size; (void)ws_size;
  const int* keys       = (const int*)d_in[0];
  const int* vals       = (const int*)d_in[1];
  const int* sentences  = (const int*)d_in[2];
  const int* sent_lens  = (const int*)d_in[3];
  const float* tau      = (const float*)d_in[4];
  const float* x_lambd  = (const float*)d_in[5];
  const float* gumbel   = (const float*)d_in[6];
  const float* drop_unif= (const float*)d_in[7];
  const float* emb      = (const float*)d_in[8];
  const float* z_emb    = (const float*)d_in[9];
  const float* enc_Wih  = (const float*)d_in[10];
  const float* enc_Whh  = (const float*)d_in[11];
  const float* enc_b    = (const float*)d_in[12];
  const float* crf_W    = (const float*)d_in[13];
  const float* crf_b    = (const float*)d_in[14];
  const float* crf_trans= (const float*)d_in[15];
  const float* init_Wh  = (const float*)d_in[16];
  const float* init_bh  = (const float*)d_in[17];
  const float* init_Wc  = (const float*)d_in[18];
  const float* init_bc  = (const float*)d_in[19];
  const float* dec_Wih  = (const float*)d_in[20];
  const float* dec_Whh  = (const float*)d_in[21];
  const float* dec_b    = (const float*)d_in[22];
  const float* attn_Wq  = (const float*)d_in[23];
  const float* attn_Wk  = (const float*)d_in[24];
  const float* attn_v   = (const float*)d_in[25];
  const float* attn_out_W=(const float*)d_in[26];
  const float* attn_out_b=(const float*)d_in[27];
  const float* pz_W     = (const float*)d_in[28];
  const float* pz_b     = (const float*)d_in[29];
  const float* int_W    = (const float*)d_in[30];
  const float* int_b    = (const float*)d_in[31];
  const float* out_W    = (const float*)d_in[32];
  const float* out_b    = (const float*)d_in[33];
  const float* cattn_Wq = (const float*)d_in[34];
  const float* cattn_Wk = (const float*)d_in[35];
  const float* cattn_v  = (const float*)d_in[36];
  const float* g_W      = (const float*)d_in[37];
  const float* g_b      = (const float*)d_in[38];
  float* out = (float*)d_out;

  size_t off = 0;
  auto alloc = [&](size_t bytes) -> char* {
    off = (off + 255) & ~(size_t)255;
    char* p = (char*)d_ws + off;
    off += bytes;
    return p;
  };
  float* sent_emb        = (float*)alloc((size_t)B_ * T_ * D_ * 4);
  unsigned short* A_enc  = (unsigned short*)alloc((size_t)B_ * 512 * 2);
  unsigned short* A_dec  = (unsigned short*)alloc((size_t)B_ * 512 * 2);
  unsigned short* Wenc_t = (unsigned short*)alloc((size_t)D4_ * 512 * 2);
  unsigned short* Wdec_t = (unsigned short*)alloc((size_t)D4_ * 512 * 2);
  unsigned short* outWt  = (unsigned short*)alloc((size_t)V_ * 256 * 2);
  float* gates           = (float*)alloc((size_t)B_ * D4_ * 4);
  float* c_enc           = (float*)alloc((size_t)B_ * D_ * 4);
  float* enc_out         = (float*)alloc((size_t)T_ * B_ * D_ * 4);
  float* praw            = (float*)alloc((size_t)T_ * B_ * K_ * 4);
  float* phiT            = (float*)alloc((size_t)T_ * B_ * K_ * 4);
  float* alphas          = (float*)alloc((size_t)T_ * B_ * K_ * 4);
  float* zsample         = (float*)alloc((size_t)B_ * T_ * K_ * 4);
  int*   z_ids           = (int*)alloc((size_t)B_ * T_ * 4);
  unsigned short* dec_in = (unsigned short*)alloc((size_t)B_ * (T_ - 1) * D_ * 2);
  float* z_next          = (float*)alloc((size_t)B_ * (T_ - 1) * D_ * 4);
  float* kv_emb          = (float*)alloc((size_t)B_ * M_ * D_ * 4);
  float* memk            = (float*)alloc((size_t)B_ * M_ * D_ * 4);
  float* memk2           = (float*)alloc((size_t)B_ * M_ * D_ * 4);
  float* h_dec           = (float*)alloc((size_t)B_ * D_ * 4);
  float* c_dec           = (float*)alloc((size_t)B_ * D_ * 4);
  unsigned short* decint = (unsigned short*)alloc((size_t)B_ * D_ * 2);
  float* scal            = (float*)alloc((size_t)B_ * 4 * 4);
  float* rowsum          = (float*)alloc((size_t)B_ * 4);
  float* accs            = (float*)alloc(2 * 4); // [0]=ent, [1]=logprob

  // --- weight prep ---
  prep_comb_w<<<dim3(D4_ * 512 / 256), dim3(256), 0, stream>>>(enc_Wih, enc_Whh, Wenc_t);
  prep_comb_w<<<dim3(D4_ * 512 / 256), dim3(256), 0, stream>>>(dec_Wih, dec_Whh, Wdec_t);
  prep_outw<<<dim3(V_ * 256 / 256), dim3(256), 0, stream>>>(out_W, outWt);

  // --- setup ---
  gather_sent<<<dim3(B_ * T_), dim3(256), 0, stream>>>(sentences, emb, sent_emb);
  enc_init<<<dim3(B_), dim3(256), 0, stream>>>(sent_emb, A_enc, c_enc, rowsum, accs);
  kv_setup<<<dim3(B_), dim3(256), 0, stream>>>(keys, vals, emb, init_Wh, init_bh, init_Wc, init_bc,
                                               attn_Wk, cattn_Wk, kv_emb, memk, memk2, h_dec, c_dec, A_dec);

  // --- encoder LSTM (WMMA gate GEMM K=512 over [x|h]) ---
  for (int t = 0; t < T_; ++t) {
    gemm_bf16<<<dim3(D4_ / 16, 1), dim3(256), 0, stream>>>(A_enc, 512, Wenc_t, 512, gates, D4_, enc_b, 512);
    lstm_pw_enc<<<dim3(B_), dim3(256), 0, stream>>>(t, gates, c_enc, enc_out, sent_emb, A_enc);
  }

  // --- CRF potentials + forward + backward sampling ---
  phi_raw_k<<<dim3(T_ * B_ * K_ / 256), dim3(256), 0, stream>>>(enc_out, crf_W, crf_b, praw);
  phi_smooth_k<<<dim3(T_ * B_ * K_ / 256), dim3(256), 0, stream>>>(praw, phiT);
  crf_fwd<<<dim3(B_), dim3(K_), 0, stream>>>(phiT, crf_trans, sent_lens, alphas, &accs[0]);
  crf_bwd<<<dim3(B_), dim3(K_), 0, stream>>>(alphas, crf_trans, gumbel, sent_lens, tau, zsample, z_ids);

  // --- decoder inputs ---
  build_dec_inputs<<<dim3(B_ * T_), dim3(256), 0, stream>>>(zsample, z_emb, sent_emb, drop_unif,
                                                            x_lambd, dec_in, z_next, A_dec);

  // --- decoder loop ---
  for (int t = 0; t < T_ - 1; ++t) {
    gemm_bf16<<<dim3(D4_ / 16, 1), dim3(256), 0, stream>>>(A_dec, 512, Wdec_t, 512, gates, D4_, dec_b, 512);
    lstm_pw_dec<<<dim3(B_), dim3(256), 0, stream>>>(t, gates, c_dec, h_dec, dec_in, A_dec);
    dec_fused_small<<<dim3(B_), dim3(256), 0, stream>>>(t, h_dec, kv_emb, memk, memk2, keys,
        attn_Wq, attn_v, attn_out_W, attn_out_b, pz_W, pz_b, int_W, int_b,
        cattn_Wq, cattn_v, g_W, g_b, out_W, out_b, z_next, sentences, z_ids, decint, scal);
    out_proj_sumexp<<<dim3(V_ / 64), dim3(256), 0, stream>>>(decint, outWt, out_b, rowsum);
    dec_finalize<<<dim3(1), dim3(B_), 0, stream>>>(t, scal, rowsum, sent_lens, &accs[1]);
  }

  final_loss<<<dim3(1), dim3(32), 0, stream>>>(accs, sent_lens, out);
}
